// MPNN_42941083025519
// MI455X (gfx1250) — matrix operand order repaired and checked by reading, not compiled, not run on hardware
//
#include <hip/hip_runtime.h>

// Problem constants (from reference)
#define TN_   65536    // total nodes
#define TE_   131072   // total edges
#define MUP_  262144   // up-adjacency entries
#define MBD_  262144   // boundary entries
#define B_    128      // graphs
#define H_    128      // hidden

typedef __attribute__((ext_vector_type(16))) _Float16 v16h;
typedef __attribute__((ext_vector_type(8)))  float    v8f;

__device__ __forceinline__ void atomAddF(float* p, float v) {
  (void)__hip_atomic_fetch_add(p, v, __ATOMIC_RELAXED, __HIP_MEMORY_SCOPE_AGENT);
}

// ---------------------------------------------------------------------------
// A-fragment: lane (L&15) = row, kgrp = L>>4.
// halves 0..7  = K: kt*32 + kgrp*8 + (0..7)
// halves 8..15 = K: kt*32 + kgrp*8 + 16 + (0..7)
// Built from 4 coalesced float4 loads + cvt f32->f16.
// ---------------------------------------------------------------------------
__device__ __forceinline__ v16h load_a_frag(const float* __restrict__ A,
                                            size_t rowBase, int kt, int kgrp) {
  const float* p = A + rowBase + (kt << 5) + (kgrp << 3);
  float4 f0 = ((const float4*)p)[0];
  float4 f1 = ((const float4*)p)[1];
  float4 f2 = ((const float4*)(p + 16))[0];
  float4 f3 = ((const float4*)(p + 16))[1];
  v16h a;
  a[0]=(_Float16)f0.x;  a[1]=(_Float16)f0.y;  a[2]=(_Float16)f0.z;  a[3]=(_Float16)f0.w;
  a[4]=(_Float16)f1.x;  a[5]=(_Float16)f1.y;  a[6]=(_Float16)f1.z;  a[7]=(_Float16)f1.w;
  a[8]=(_Float16)f2.x;  a[9]=(_Float16)f2.y;  a[10]=(_Float16)f2.z; a[11]=(_Float16)f2.w;
  a[12]=(_Float16)f3.x; a[13]=(_Float16)f3.y; a[14]=(_Float16)f3.z; a[15]=(_Float16)f3.w;
  return a;
}

// B-fragment: weights pre-packed so each lane's 16 halves are contiguous (32 B).
__device__ __forceinline__ v16h load_b_frag(const _Float16* __restrict__ Wp,
                                            int kt, int ct, int lane, int ntiles) {
  return *(const v16h*)(Wp + ((((size_t)kt * ntiles + ct) << 5) + lane) * 16);
}

// Pack f32 W[K][N] (row-major) into fragment order: Wp[kt][ct][lane][i]
// where k = kt*32 + (i>>3)*16 + (lane>>4)*8 + (i&7), n = ct*16 + (lane&15).
__global__ void pack_w_kernel(const float* __restrict__ W, _Float16* __restrict__ Wp,
                              int K, int N) {
  int idx = blockIdx.x * blockDim.x + threadIdx.x;
  if (idx >= K * N) return;
  int i    = idx & 15;
  int lane = (idx >> 4) & 31;
  int rest = idx >> 9;
  int ntiles = N >> 4;
  int ct = rest % ntiles;
  int kt = rest / ntiles;
  int k = (kt << 5) + ((i >> 3) << 4) + ((lane >> 4) << 3) + (i & 7);
  int n = (ct << 4) + (lane & 15);
  Wp[idx] = (_Float16)W[(size_t)k * N + n];
}

// ---------------------------------------------------------------------------
// Fused WMMA GEMM: C = epilogue(A1@W1 [+ A2@W2] + bias)
// One wave owns a 16x16 tile; block = 8 waves = 16 rows x 128 cols.
// EPI: 0 = relu, 1 = sigmoid-gate (C = sigmoid(acc+b) * C_old), 2 = none.
// INPLACE: preload ALL A and B fragments (K==128), barrier, then a pure
// back-to-back WMMA accumulate chain (no VALU between WMMAs -> no hazard
// NOPs, single load-wait at the barrier). Safe because with N==128 each
// block exclusively owns rows r0..r0+15.
// ---------------------------------------------------------------------------
template <int HAS_A2, int EPI, int INPLACE>
__global__ __launch_bounds__(256) void wmma_gemm_kernel(
    const float* __restrict__ A1, const float* __restrict__ A2,
    const _Float16* __restrict__ W1, const _Float16* __restrict__ W2,
    const float* __restrict__ bias, float* __restrict__ C,
    int M, int N, int K, int lda, int ldc) {
  (void)M;
  const int wave = threadIdx.x >> 5;
  const int lane = threadIdx.x & 31;
  const int r0 = blockIdx.x << 4;
  const int ct = blockIdx.y * 8 + wave;
  const int c0 = ct << 4;
  const int kgrp = lane >> 4;
  const int ntiles = N >> 4;
  const size_t rowBase = (size_t)(r0 + (lane & 15)) * lda;
  const int col = c0 + (lane & 15);

  v8f acc = {};
  float gate[8] = {0.f, 0.f, 0.f, 0.f, 0.f, 0.f, 0.f, 0.f};

  if (INPLACE) {
    // K == 128 on this path
    v16h a1[4], a2[4], b1[4], b2[4];
#pragma unroll
    for (int kt = 0; kt < 4; ++kt) {
      a1[kt] = load_a_frag(A1, rowBase, kt, kgrp);
      if (HAS_A2) a2[kt] = load_a_frag(A2, rowBase, kt, kgrp);
    }
#pragma unroll
    for (int kt = 0; kt < 4; ++kt) {
      b1[kt] = load_b_frag(W1, kt, ct, lane, ntiles);
      if (HAS_A2) b2[kt] = load_b_frag(W2, kt, ct, lane, ntiles);
    }
    if (EPI == 1) {
#pragma unroll
      for (int v = 0; v < 8; ++v)
        gate[v] = C[(size_t)(r0 + v + (kgrp << 3)) * ldc + col];
    }
    __syncthreads();  // waits all loads; orders this block's reads before stores
#pragma unroll
    for (int kt = 0; kt < 4; ++kt) {
      acc = __builtin_amdgcn_wmma_f32_16x16x32_f16(false, a1[kt], false, b1[kt],
                                                   (short)0, acc, false, false);
      if (HAS_A2)
        acc = __builtin_amdgcn_wmma_f32_16x16x32_f16(false, a2[kt], false, b2[kt],
                                                     (short)0, acc, false, false);
    }
  } else {
    const int KT = K >> 5;
    for (int kt = 0; kt < KT; ++kt) {
      v16h a = load_a_frag(A1, rowBase, kt, kgrp);
      v16h b = load_b_frag(W1, kt, ct, lane, ntiles);
      acc = __builtin_amdgcn_wmma_f32_16x16x32_f16(false, a, false, b,
                                                   (short)0, acc, false, false);
      if (HAS_A2) {
        v16h a2v = load_a_frag(A2, rowBase, kt, kgrp);
        v16h b2v = load_b_frag(W2, kt, ct, lane, ntiles);
        acc = __builtin_amdgcn_wmma_f32_16x16x32_f16(false, a2v, false, b2v,
                                                     (short)0, acc, false, false);
      }
    }
  }

  const float bv = bias[col];
#pragma unroll
  for (int v = 0; v < 8; ++v) {
    float x = acc[v] + bv;
    if (EPI == 0)      x = fmaxf(x, 0.f);
    else if (EPI == 1) x = gate[v] * (1.f / (1.f + __expf(-x)));
    C[(size_t)(r0 + v + (kgrp << 3)) * ldc + col] = x;
  }
}

// ---------------------------------------------------------------------------
// Gather / scatter kernels: 1 wave per entry, float4 per lane (512 B rows).
// ---------------------------------------------------------------------------
__global__ void embed_kernel(const int* __restrict__ types, const float4* __restrict__ emb,
                             float4* __restrict__ out, int rows) {
  int t = blockIdx.x * blockDim.x + threadIdx.x;
  int r = t >> 5, lane = t & 31;
  if (r >= rows) return;
  out[(size_t)r * 32 + lane] = emb[(size_t)types[r] * 32 + lane];
}

__global__ void agg_up_kernel(const float4* __restrict__ xn, const float4* __restrict__ xe,
                              const int* __restrict__ up_i, const int* __restrict__ up_j,
                              const int* __restrict__ up_attr,
                              float* __restrict__ agg_n, int m) {
  int t = blockIdx.x * blockDim.x + threadIdx.x;
  int e = t >> 5, lane = t & 31;
  if (e >= m) return;
  float4 v = xn[(size_t)up_j[e] * 32 + lane];
  float4 w = xe[(size_t)up_attr[e] * 32 + lane];
  float* dst = agg_n + (size_t)up_i[e] * 128 + lane * 4;
  atomAddF(dst + 0, v.x + w.x);
  atomAddF(dst + 1, v.y + w.y);
  atomAddF(dst + 2, v.z + w.z);
  atomAddF(dst + 3, v.w + w.w);
}

__global__ void agg_bd_kernel(const float4* __restrict__ xn, const int* __restrict__ bd_e,
                              const int* __restrict__ bd_n,
                              float* __restrict__ agg_e, int m) {
  int t = blockIdx.x * blockDim.x + threadIdx.x;
  int e = t >> 5, lane = t & 31;
  if (e >= m) return;
  float4 v = xn[(size_t)bd_n[e] * 32 + lane];
  float* dst = agg_e + (size_t)bd_e[e] * 128 + lane * 4;
  atomAddF(dst + 0, v.x);
  atomAddF(dst + 1, v.y);
  atomAddF(dst + 2, v.z);
  atomAddF(dst + 3, v.w);
}

__global__ void zero_kernel(float4* __restrict__ p, int n4) {
  int i = blockIdx.x * blockDim.x + threadIdx.x;
  if (i < n4) p[i] = make_float4(0.f, 0.f, 0.f, 0.f);
}

// Per-graph sum pooling (mask is all-ones: scatter indices are a bijection).
__global__ void pool_kernel(const float* __restrict__ X, float* __restrict__ P, int rows) {
  int b = blockIdx.x, h = threadIdx.x;  // 128 threads = 128 cols
  const float* p = X + (size_t)b * rows * 128 + h;
  float s = 0.f;
  for (int r = 0; r < rows; ++r) s += p[(size_t)r * 128];
  P[b * 128 + h] = s;
}

// ---------------------------------------------------------------------------
extern "C" void kernel_launch(void* const* d_in, const int* in_sizes, int n_in,
                              void* d_out, int out_size, void* d_ws, size_t ws_size,
                              hipStream_t stream) {
  (void)in_sizes; (void)n_in; (void)out_size;
  const int*   node_types = (const int*)d_in[0];
  const int*   edge_types = (const int*)d_in[1];
  const int*   up_i    = (const int*)d_in[2];
  const int*   up_j    = (const int*)d_in[3];
  const int*   up_attr = (const int*)d_in[4];
  const int*   bd_e    = (const int*)d_in[5];
  const int*   bd_n    = (const int*)d_in[6];
  const float* atom_emb = (const float*)d_in[7];
  const float* bond_emb = (const float*)d_in[8];
  const float* Wn_self = (const float*)d_in[9];
  const float* Wn_agg  = (const float*)d_in[10];
  const float* bn_b    = (const float*)d_in[11];
  const float* We_self = (const float*)d_in[12];
  const float* We_agg  = (const float*)d_in[13];
  const float* be_b    = (const float*)d_in[14];
  const float* Wpx     = (const float*)d_in[15];
  const float* Wpr     = (const float*)d_in[16];
  const float* bp      = (const float*)d_in[17];
  const float* lin1_w  = (const float*)d_in[18];
  const float* lin1_b  = (const float*)d_in[19];
  const float* lin2_w  = (const float*)d_in[20];
  const float* lin2_b  = (const float*)d_in[21];

  // ---- workspace layout ----
  char* wsp = (char*)d_ws;
  size_t off = 0;
  auto ALLOC = [&](size_t bytes) -> char* {
    char* p = wsp + off;
    off = (off + bytes + 255) & ~(size_t)255;
    return p;
  };
  float* xn    = (float*)ALLOC((size_t)TN_ * 128 * 4);
  float* xe    = (float*)ALLOC((size_t)TE_ * 128 * 4);
  float* agg_n = (float*)ALLOC((size_t)TN_ * 128 * 4);
  float* agg_e = (float*)ALLOC((size_t)TE_ * 128 * 4);
  float* pn    = (float*)ALLOC(128 * 128 * 4);
  float* pe    = (float*)ALLOC(128 * 128 * 4);
  float* hcat  = (float*)ALLOC(128 * 512 * 4);
  _Float16* Wns_p  = (_Float16*)ALLOC(3 * 128 * 128 * 2);
  _Float16* Wna_p  = (_Float16*)ALLOC(3 * 128 * 128 * 2);
  _Float16* Wes_p  = (_Float16*)ALLOC(3 * 128 * 128 * 2);
  _Float16* Wea_p  = (_Float16*)ALLOC(3 * 128 * 128 * 2);
  _Float16* Wpx_p  = (_Float16*)ALLOC(128 * 128 * 2);
  _Float16* Wpr_p  = (_Float16*)ALLOC(128 * 128 * 2);
  _Float16* lin1_p = (_Float16*)ALLOC(2 * 128 * 256 * 2);
  _Float16* lin2_p = (_Float16*)ALLOC(512 * 128 * 2);
  if (ws_size < off) return;  // insufficient scratch

  const int TPB = 256;
  // ---- pack weights into WMMA B-fragment order (fp16) ----
  for (int l = 0; l < 3; ++l) {
    pack_w_kernel<<<64, TPB, 0, stream>>>(Wn_self + l * 16384, Wns_p + l * 16384, 128, 128);
    pack_w_kernel<<<64, TPB, 0, stream>>>(Wn_agg  + l * 16384, Wna_p + l * 16384, 128, 128);
    pack_w_kernel<<<64, TPB, 0, stream>>>(We_self + l * 16384, Wes_p + l * 16384, 128, 128);
    pack_w_kernel<<<64, TPB, 0, stream>>>(We_agg  + l * 16384, Wea_p + l * 16384, 128, 128);
  }
  pack_w_kernel<<<64, TPB, 0, stream>>>(Wpx, Wpx_p, 128, 128);
  pack_w_kernel<<<64, TPB, 0, stream>>>(Wpr, Wpr_p, 128, 128);
  pack_w_kernel<<<128, TPB, 0, stream>>>(lin1_w,         lin1_p,         128, 256);
  pack_w_kernel<<<128, TPB, 0, stream>>>(lin1_w + 32768, lin1_p + 32768, 128, 256);
  pack_w_kernel<<<256, TPB, 0, stream>>>(lin2_w, lin2_p, 512, 128);

  // ---- init: xn = atom_emb[node_types]; xe = bond_emb[edge_types] + seg(xn[bd_n], bd_e)
  embed_kernel<<<(TN_ * 32) / TPB, TPB, 0, stream>>>(node_types, (const float4*)atom_emb, (float4*)xn, TN_);
  embed_kernel<<<(TE_ * 32) / TPB, TPB, 0, stream>>>(edge_types, (const float4*)bond_emb, (float4*)xe, TE_);
  agg_bd_kernel<<<(MBD_ * 32) / TPB, TPB, 0, stream>>>((const float4*)xn, bd_e, bd_n, xe, MBD_);

  // ---- 3 message-passing layers ----
  for (int l = 0; l < 3; ++l) {
    zero_kernel<<<(TN_ * 32) / TPB, TPB, 0, stream>>>((float4*)agg_n, TN_ * 32);
    zero_kernel<<<(TE_ * 32) / TPB, TPB, 0, stream>>>((float4*)agg_e, TE_ * 32);
    agg_up_kernel<<<(MUP_ * 32) / TPB, TPB, 0, stream>>>((const float4*)xn, (const float4*)xe,
                                                         up_i, up_j, up_attr, agg_n, MUP_);
    agg_bd_kernel<<<(MBD_ * 32) / TPB, TPB, 0, stream>>>((const float4*)xn, bd_e, bd_n, agg_e, MBD_);
    // xn = relu(xn@Wn_self + agg_n@Wn_agg + b)   (in place, K=2x128 fused)
    wmma_gemm_kernel<1, 0, 1><<<dim3(TN_ / 16, 1), 256, 0, stream>>>(
        xn, agg_n, Wns_p + l * 16384, Wna_p + l * 16384, bn_b + l * 128, xn,
        TN_, 128, 128, 128, 128);
    wmma_gemm_kernel<1, 0, 1><<<dim3(TE_ / 16, 1), 256, 0, stream>>>(
        xe, agg_e, Wes_p + l * 16384, Wea_p + l * 16384, be_b + l * 128, xe,
        TE_, 128, 128, 128, 128);
  }

  // ---- gated pooling weight: recompute ref activations into agg buffers ----
  embed_kernel<<<(TN_ * 32) / TPB, TPB, 0, stream>>>(node_types, (const float4*)atom_emb, (float4*)agg_n, TN_);
  embed_kernel<<<(TE_ * 32) / TPB, TPB, 0, stream>>>(edge_types, (const float4*)bond_emb, (float4*)agg_e, TE_);
  agg_bd_kernel<<<(MBD_ * 32) / TPB, TPB, 0, stream>>>((const float4*)agg_n, bd_e, bd_n, agg_e, MBD_);
  // x = sigmoid(x@Wpx + ref@Wpr + bp) * x   (in place)
  wmma_gemm_kernel<1, 1, 1><<<dim3(TN_ / 16, 1), 256, 0, stream>>>(
      xn, agg_n, Wpx_p, Wpr_p, bp, xn, TN_, 128, 128, 128, 128);
  wmma_gemm_kernel<1, 1, 1><<<dim3(TE_ / 16, 1), 256, 0, stream>>>(
      xe, agg_e, Wpx_p, Wpr_p, bp, xe, TE_, 128, 128, 128, 128);

  // ---- per-graph sum pooling ----
  pool_kernel<<<B_, 128, 0, stream>>>(xn, pn, 512);
  pool_kernel<<<B_, 128, 0, stream>>>(xe, pe, 1024);

  // ---- head: h0|h1 = relu(p @ lin1 + b) -> concat -> @ lin2 + b ----
  wmma_gemm_kernel<0, 0, 0><<<dim3(8, 2), 256, 0, stream>>>(
      pn, nullptr, lin1_p, nullptr, lin1_b, hcat, 128, 256, 128, 128, 512);
  wmma_gemm_kernel<0, 0, 0><<<dim3(8, 2), 256, 0, stream>>>(
      pe, nullptr, lin1_p + 32768, nullptr, lin1_b + 256, hcat + 256, 128, 256, 128, 128, 512);
  wmma_gemm_kernel<0, 2, 0><<<dim3(8, 1), 256, 0, stream>>>(
      hcat, nullptr, lin2_p, nullptr, lin2_b, (float*)d_out, 128, 128, 512, 512, 128);
}
